// Encoder_74388833567042
// MI455X (gfx1250) — compile-verified, matrix operand order checked by reading, not run on hardware
//
#include <hip/hip_runtime.h>
#include <math.h>

// ---------------- problem constants (match reference) ----------------
#define S_LEN   1024
#define BATCH   4
#define DMODEL  1024
#define NHEAD   16
#define DHEAD   64
#define FDIM    4096
#define NLAYER  6
#define MTOT    (BATCH * S_LEN)   // 4096 rows for all GEMMs

// ---------------- CDNA5 WMMA plumbing ----------------
typedef __attribute__((ext_vector_type(16))) __bf16 v16bf;
typedef __attribute__((ext_vector_type(8)))  float  v8f;
typedef __attribute__((ext_vector_type(4)))  unsigned int v4u;
typedef __attribute__((ext_vector_type(8)))  unsigned int v8u;

union BF16x16 { v16bf v; unsigned short u[16]; };

__device__ __forceinline__ unsigned short f2bf(float x) {
  unsigned int u = __float_as_uint(x);
  u += 0x7FFFu + ((u >> 16) & 1u);          // round-to-nearest-even
  return (unsigned short)(u >> 16);
}

// Build a 16x32 bf16 WMMA operand from an LDS tile.
// A matrix: tile row-major A[m][k], ld = row stride (elements).
// B matrix: pass the *transposed* tile Bt[n][k]; mapping mirrors A.
// Layout per ISA 7.12.2 (16-bit A 16x32): lane = {m = lane&15, g = lane>>4},
// vgpr v holds K pair kbase..kbase+1, kbase = (v<4?0:16) + 8*g + 2*(v&3).
__device__ __forceinline__ v16bf load_frag(const unsigned short* p, int ld) {
  const int lane = threadIdx.x & 31;
  const int m = lane & 15;
  const int g = lane >> 4;
  BF16x16 r;
#pragma unroll
  for (int v = 0; v < 8; ++v) {
    const int k = ((v & 4) << 2) + g * 8 + ((v & 3) << 1);
    unsigned int pair;
    __builtin_memcpy(&pair, p + m * ld + k, 4);
    r.u[2 * v]     = (unsigned short)(pair & 0xFFFFu);
    r.u[2 * v + 1] = (unsigned short)(pair >> 16);
  }
  return r.v;
}

__device__ __forceinline__ v8f wmma_bf16(v16bf a, v16bf b, v8f c) {
  return __builtin_amdgcn_wmma_f32_16x16x32_bf16(false, a, false, b,
                                                 (short)0, c, false, false);
}

// ---------------- TDM: async 2D bf16 tile load, global -> LDS ---------------
// D# per CDNA5 ISA: group0 {count=1, lds_addr,
// global_addr[56:0], type=2}; group1 {data_size=2B, tensor/tile dims, stride}.
// 2D tensor => only groups 0..1 (VADDR2/3 omitted). Tracked by TENSORcnt.
__device__ __forceinline__ void tdm_load_2d(unsigned lds_off,
                                            const unsigned short* gptr,
                                            unsigned row_stride_elems,
                                            unsigned tile_k,      // dim0 (contig)
                                            unsigned tile_rows) { // dim1
  const unsigned long long ga = (unsigned long long)(size_t)gptr;
  v4u g0;
  g0[0] = 1u;                                          // count=1 (valid D#)
  g0[1] = lds_off;                                     // lds_addr (bytes)
  g0[2] = (unsigned)(ga & 0xFFFFFFFFu);                // global_addr[31:0]
  g0[3] = (unsigned)((ga >> 32) & 0x01FFFFFFu) | (2u << 30); // addr[56:32]|type=2
  v8u g1;
  g1[0] = 1u << 16;                                    // data_size = 2 bytes
  g1[1] = (tile_k & 0xFFFFu) << 16;                    // tensor_dim0[15:0]
  g1[2] = ((tile_k >> 16) & 0xFFFFu) | ((tile_rows & 0xFFFFu) << 16); // td0 hi|td1 lo
  g1[3] = ((tile_rows >> 16) & 0xFFFFu) | ((tile_k & 0xFFFFu) << 16); // td1 hi|tile_dim0
  g1[4] = tile_rows & 0xFFFFu;                         // tile_dim1 (tile_dim2=0)
  g1[5] = row_stride_elems;                            // tensor_dim0_stride[31:0]
  g1[6] = 0u;
  g1[7] = 0u;
  asm volatile("tensor_load_to_lds %0, %1" :: "s"(g0), "s"(g1) : "memory");
}

__device__ __forceinline__ void wait_dscnt0() {
  asm volatile("s_wait_dscnt 0x0" ::: "memory");
}

// T5 relative position bucket (non-causal, NUM_BUCKETS=32, MAX_DIST=256)
__device__ __forceinline__ int t5_bucket(int q, int k) {
  int n = q - k;
  int ret = 0;
  if (n < 0) { ret = 16; n = -n; }
  if (n < 8) return ret + n;
  int v = 8 + (int)(__logf((float)n * 0.125f) * (8.0f / __logf(32.0f)));
  return ret + (v < 15 ? v : 15);
}

// ---------------- one-time weight convert + transpose (fp32[K][N]->bf16[N][K])
__global__ __launch_bounds__(256) void convert_transpose_kernel(
    const float* __restrict__ W, unsigned short* __restrict__ Wt, int K, int N) {
  __shared__ float tile[64][65];
  const int k0 = blockIdx.y * 64, n0 = blockIdx.x * 64;
  for (int i = threadIdx.x; i < 64 * 64; i += 256) {
    const int r = i >> 6, c = i & 63;
    tile[r][c] = W[(size_t)(k0 + r) * N + n0 + c];
  }
  __syncthreads();
  for (int i = threadIdx.x; i < 64 * 32; i += 256) {   // packed u32 writes
    const int r = i >> 5, c2 = i & 31;
    const unsigned lo = f2bf(tile[2 * c2][r]);
    const unsigned hi = f2bf(tile[2 * c2 + 1][r]);
    *(unsigned*)(Wt + (size_t)(n0 + r) * K + k0 + 2 * c2) = lo | (hi << 16);
  }
}

// ---------------- embedding (fp32 + bf16 copies) ----------------
__global__ __launch_bounds__(256) void embed_kernel(
    const int* __restrict__ src, const float* __restrict__ tok,
    const float* __restrict__ pos, float* __restrict__ x,
    unsigned short* __restrict__ xb) {
  const int bs = blockIdx.x;
  const int s  = bs & (S_LEN - 1);
  const int t  = src[bs];
#pragma unroll
  for (int i = 0; i < DMODEL / 256; ++i) {
    const int d = threadIdx.x + i * 256;
    const float v = tok[(size_t)t * DMODEL + d] * 32.0f + pos[(size_t)s * DMODEL + d];
    x[(size_t)bs * DMODEL + d]  = v;
    xb[(size_t)bs * DMODEL + d] = f2bf(v);
  }
}

// ---------------- residual + layernorm (fp32 in-place + bf16 copy) ----------
__global__ __launch_bounds__(256) void ln_res_kernel(
    float* __restrict__ x, const float* __restrict__ r,
    const float* __restrict__ g, const float* __restrict__ b,
    unsigned short* __restrict__ xb) {
  const int row = blockIdx.x;
  const size_t base = (size_t)row * DMODEL;
  float v0[DMODEL / 256];
  float s = 0.0f, sq = 0.0f;
#pragma unroll
  for (int i = 0; i < DMODEL / 256; ++i) {
    const int d = threadIdx.x + i * 256;
    const float v = x[base + d] + r[base + d];
    v0[i] = v; s += v; sq += v * v;
  }
#pragma unroll
  for (int m = 16; m >= 1; m >>= 1) {
    s  += __shfl_xor(s, m, 32);
    sq += __shfl_xor(sq, m, 32);
  }
  __shared__ float ss[8], ssq[8];
  const int wave = threadIdx.x >> 5, lane = threadIdx.x & 31;
  if (lane == 0) { ss[wave] = s; ssq[wave] = sq; }
  __syncthreads();
  s = 0.0f; sq = 0.0f;
#pragma unroll
  for (int w = 0; w < 8; ++w) { s += ss[w]; sq += ssq[w]; }
  const float mu  = s * (1.0f / DMODEL);
  const float var = sq * (1.0f / DMODEL) - mu * mu;
  const float inv = rsqrtf(var + 1e-5f);
#pragma unroll
  for (int i = 0; i < DMODEL / 256; ++i) {
    const int d = threadIdx.x + i * 256;
    const float o = (v0[i] - mu) * inv * g[d] + b[d];
    x[base + d]  = o;
    xb[base + d] = f2bf(o);
  }
}

// ---------------- TDM double-buffered bf16 WMMA GEMM ----------------
// C = act(A @ B + bias); A: bf16 [M][K]; Bt: bf16 [N][K] (pre-transposed).
// flags: 1=relu, 2=write fp32 C, 4=write bf16 Cb row-major,
//        8=write bf16 Cb in V-layout [B][H][DH][S].
// Block tile 128x128, staged depth BK=64 (two WMMA K-chunks per barrier pair):
// 16 v_wmma per wave per iteration.
#define BM 128
#define BN 128
#define BK 64

__global__ __launch_bounds__(256) void gemm_bf16_tdm_kernel(
    const unsigned short* __restrict__ A, const unsigned short* __restrict__ Bt,
    const float* __restrict__ bias, float* __restrict__ C,
    unsigned short* __restrict__ Cb, int M, int N, int K, int flags) {
  __shared__ __align__(16) unsigned short As[2][BM][BK];   // 2 x 16 KB
  __shared__ __align__(16) unsigned short Bs[2][BN][BK];   // 2 x 16 KB

  const int tid  = threadIdx.x;
  const int wave = tid >> 5;
  const int lane = tid & 31;
  const int bm = blockIdx.y * BM;
  const int bn = blockIdx.x * BN;
  const int wm = (wave >> 1) * 32;   // 4 wave-rows of 32
  const int wn = (wave & 1) * 64;    // 2 wave-cols of 64

  v8f acc[2][4] = {};

  const int ksteps = K / BK;
  if (wave == 0) {
    tdm_load_2d((unsigned)(size_t)&As[0][0][0], A + (size_t)bm * K, K, BK, BM);
    tdm_load_2d((unsigned)(size_t)&Bs[0][0][0], Bt + (size_t)bn * K, K, BK, BN);
  }

  for (int kc = 0; kc < ksteps; ++kc) {
    const int cur = kc & 1;
    if (wave == 0) {
      if (kc + 1 < ksteps) {
        const int k0 = (kc + 1) * BK;
        tdm_load_2d((unsigned)(size_t)&As[cur ^ 1][0][0],
                    A + (size_t)bm * K + k0, K, BK, BM);
        tdm_load_2d((unsigned)(size_t)&Bs[cur ^ 1][0][0],
                    Bt + (size_t)bn * K + k0, K, BK, BN);
        __builtin_amdgcn_s_wait_tensorcnt((short)2);  // current pair retired
      } else {
        __builtin_amdgcn_s_wait_tensorcnt((short)0);
      }
    }
    __syncthreads();

#pragma unroll
    for (int kk = 0; kk < 2; ++kk) {
      v16bf af[2], bfr[4];
      af[0] = load_frag(&As[cur][wm][kk * 32], BK);
      af[1] = load_frag(&As[cur][wm + 16][kk * 32], BK);
#pragma unroll
      for (int ni = 0; ni < 4; ++ni)
        bfr[ni] = load_frag(&Bs[cur][wn + ni * 16][kk * 32], BK);
#pragma unroll
      for (int mi = 0; mi < 2; ++mi)
#pragma unroll
        for (int ni = 0; ni < 4; ++ni)
          acc[mi][ni] = wmma_bf16(af[mi], bfr[ni], acc[mi][ni]);
    }
    __syncthreads();
  }

  const int n = lane & 15;
  const int g = lane >> 4;
#pragma unroll
  for (int mi = 0; mi < 2; ++mi) {
#pragma unroll
    for (int ni = 0; ni < 4; ++ni) {
#pragma unroll
      for (int v = 0; v < 8; ++v) {
        const int m  = bm + wm + mi * 16 + g * 8 + v;
        const int nn = bn + wn + ni * 16 + n;
        float o = acc[mi][ni][v] + bias[nn];
        if (flags & 1) o = fmaxf(o, 0.0f);
        if (flags & 2) C[(size_t)m * N + nn] = o;
        if (flags & 4) Cb[(size_t)m * N + nn] = f2bf(o);
        if (flags & 8) {  // V layout: [B][H][DH][S]
          const int b_ = m >> 10, s_ = m & 1023;
          Cb[(((size_t)b_ * NHEAD + (nn >> 6)) * DHEAD + (nn & 63)) * S_LEN + s_] =
              f2bf(o);
        }
      }
    }
  }
}

// ---------------- flash attention: one wave per (b, h, 16-query tile) -------
// Q,K bf16 [B*S][D]; V bf16 [B][H][DH][S]; output AO bf16 [B*S][D].
// K/V chunks stream in via TDM, double-buffered; DMA overlaps softmax VALU.
__global__ __launch_bounds__(32) void attn_flash_kernel(
    const unsigned short* __restrict__ Q, const unsigned short* __restrict__ K,
    const unsigned short* __restrict__ V, const int* __restrict__ mask,
    const float* __restrict__ relb, unsigned short* __restrict__ O) {
  __shared__ __align__(16) unsigned short Qs[16][64];
  __shared__ __align__(16) unsigned short Ks[2][32][64];
  __shared__ __align__(16) unsigned short Vt[2][64][32];  // [d][key]
  __shared__ __align__(16) unsigned short Pt[16][32];
  __shared__ float Sc[16][32];
  __shared__ float rm[16], rs[16], fac[16];

  const int lane = threadIdx.x;
  const int g    = lane >> 4;
  const int qt   = blockIdx.x * 16;
  const int h    = blockIdx.y;
  const int b    = blockIdx.z;

  const unsigned short* qb = Q + ((size_t)b * S_LEN + qt) * DMODEL + h * DHEAD;
  const unsigned short* kb = K + ((size_t)b * S_LEN) * DMODEL + h * DHEAD;
  const unsigned short* vb = V + (((size_t)b * NHEAD + h) * DHEAD) * S_LEN;

  // prologue: Q tile + first K/V chunk via TDM
  tdm_load_2d((unsigned)(size_t)&Qs[0][0],    qb, DMODEL, 64, 16);
  tdm_load_2d((unsigned)(size_t)&Ks[0][0][0], kb, DMODEL, 64, 32);
  tdm_load_2d((unsigned)(size_t)&Vt[0][0][0], vb, S_LEN, 32, 64);

  if (lane < 16) { rm[lane] = -3.0e38f; rs[lane] = 0.0f; }
  __builtin_amdgcn_s_wait_tensorcnt((short)2);   // Q tile complete
  __syncthreads();

  const v16bf aq0 = load_frag(&Qs[0][0], 64);
  const v16bf aq1 = load_frag(&Qs[0][32], 64);

  v8f o[4] = {};

  for (int kc = 0; kc < S_LEN / 32; ++kc) {
    const int cur = kc & 1;
    const int kbase = kc * 32;
    if (kc + 1 < S_LEN / 32) {
      // previous-buffer fragment reads must retire before DMA overwrites it
      wait_dscnt0();
      tdm_load_2d((unsigned)(size_t)&Ks[cur ^ 1][0][0],
                  kb + (size_t)(kbase + 32) * DMODEL, DMODEL, 64, 32);
      tdm_load_2d((unsigned)(size_t)&Vt[cur ^ 1][0][0],
                  vb + (kbase + 32), S_LEN, 32, 64);
      __builtin_amdgcn_s_wait_tensorcnt((short)2); // current chunk complete
    } else {
      __builtin_amdgcn_s_wait_tensorcnt((short)0);
    }

    // scores = Q @ K^T / sqrt(DH) ; mask ; + rel_bias * REL_SCALE
#pragma unroll
    for (int t = 0; t < 2; ++t) {
      v8f s = {};
      s = wmma_bf16(aq0, load_frag(&Ks[cur][t * 16][0], 64), s);
      s = wmma_bf16(aq1, load_frag(&Ks[cur][t * 16][32], 64), s);
      const int n = lane & 15;
      const int kcol = kbase + t * 16 + n;
      const int masked = (mask[b * S_LEN + kcol] == 0);
#pragma unroll
      for (int v = 0; v < 8; ++v) {
        const int m = g * 8 + v;
        float e = masked ? -1.0e10f : (s[v] * 0.125f);
        e += relb[t5_bucket(qt + m, kcol) * NHEAD + h] * 8.0f;
        Sc[m][t * 16 + n] = e;
      }
    }
    __syncthreads();

    if (lane < 16) {                                    // online softmax
      const float mprev = rm[lane];
      float mnew = mprev;
#pragma unroll
      for (int c = 0; c < 32; ++c) mnew = fmaxf(mnew, Sc[lane][c]);
      const float f = __expf(mprev - mnew);
      float sum = rs[lane] * f;
#pragma unroll
      for (int c = 0; c < 32; ++c) {
        const float e = __expf(Sc[lane][c] - mnew);
        Pt[lane][c] = f2bf(e);
        sum += e;
      }
      rm[lane] = mnew; rs[lane] = sum; fac[lane] = f;
    }
    __syncthreads();

    float fv[8];
#pragma unroll
    for (int v = 0; v < 8; ++v) fv[v] = fac[g * 8 + v];

    const v16bf ap = load_frag(&Pt[0][0], 32);
#pragma unroll
    for (int nt = 0; nt < 4; ++nt) {
#pragma unroll
      for (int v = 0; v < 8; ++v) o[nt][v] *= fv[v];
      o[nt] = wmma_bf16(ap, load_frag(&Vt[cur][nt * 16][0], 32), o[nt]);
    }
  }

  const int n = lane & 15;
  float inv[8];
#pragma unroll
  for (int v = 0; v < 8; ++v) inv[v] = 1.0f / rs[g * 8 + v];
#pragma unroll
  for (int nt = 0; nt < 4; ++nt) {
#pragma unroll
    for (int v = 0; v < 8; ++v) {
      const int m = g * 8 + v;
      O[((size_t)b * S_LEN + qt + m) * DMODEL + h * DHEAD + nt * 16 + n] =
          f2bf(o[nt][v] * inv[v]);
    }
  }
}

// ---------------- driver ----------------
extern "C" void kernel_launch(void* const* d_in, const int* in_sizes, int n_in,
                              void* d_out, int out_size, void* d_ws, size_t ws_size,
                              hipStream_t stream) {
  (void)in_sizes; (void)n_in; (void)out_size; (void)ws_size;

  const int*   src      = (const int*)  d_in[0];
  const int*   src_mask = (const int*)  d_in[1];
  const float* tok_emb  = (const float*)d_in[2];
  const float* pos_emb  = (const float*)d_in[3];
  const float* Wq = (const float*)d_in[4];
  const float* bq = (const float*)d_in[5];
  const float* Wk = (const float*)d_in[6];
  const float* bk = (const float*)d_in[7];
  const float* Wv = (const float*)d_in[8];
  const float* bv = (const float*)d_in[9];
  const float* Wo = (const float*)d_in[10];
  const float* bo = (const float*)d_in[11];
  const float* rel_bias = (const float*)d_in[12];
  const float* ln1_g = (const float*)d_in[13];
  const float* ln1_b = (const float*)d_in[14];
  const float* W1 = (const float*)d_in[15];
  const float* b1 = (const float*)d_in[16];
  const float* W2 = (const float*)d_in[17];
  const float* b2 = (const float*)d_in[18];
  const float* ln2_g = (const float*)d_in[19];
  const float* ln2_b = (const float*)d_in[20];

  const size_t ND = (size_t)MTOT * DMODEL;   // 4  M elements
  const size_t NF = (size_t)MTOT * FDIM;     // 16 M elements
  const size_t DD = (size_t)DMODEL * DMODEL;
  const size_t DF = (size_t)DMODEL * FDIM;

  char* w = (char*)d_ws;
  float*          X   = (float*)w;          w += ND * 4;
  float*          T1  = (float*)w;          w += ND * 4;
  unsigned short* Xb  = (unsigned short*)w; w += ND * 2;
  unsigned short* Qb  = (unsigned short*)w; w += ND * 2;
  unsigned short* Kb  = (unsigned short*)w; w += ND * 2;
  unsigned short* Vb  = (unsigned short*)w; w += ND * 2;
  unsigned short* AOb = (unsigned short*)w; w += ND * 2;
  unsigned short* FFb = (unsigned short*)w; w += NF * 2;
  unsigned short* WT  = (unsigned short*)w;               // 72M bf16 elems

  const size_t LW = 4 * DD + 2 * DF;        // bf16 elems per layer

  // one-time: convert + transpose all weights to bf16 [N][K]
  for (int l = 0; l < NLAYER; ++l) {
    unsigned short* base = WT + (size_t)l * LW;
    convert_transpose_kernel<<<dim3(16, 16), 256, 0, stream>>>(Wq + l * DD, base, DMODEL, DMODEL);
    convert_transpose_kernel<<<dim3(16, 16), 256, 0, stream>>>(Wk + l * DD, base + DD, DMODEL, DMODEL);
    convert_transpose_kernel<<<dim3(16, 16), 256, 0, stream>>>(Wv + l * DD, base + 2 * DD, DMODEL, DMODEL);
    convert_transpose_kernel<<<dim3(16, 16), 256, 0, stream>>>(Wo + l * DD, base + 3 * DD, DMODEL, DMODEL);
    convert_transpose_kernel<<<dim3(64, 16), 256, 0, stream>>>(W1 + l * DF, base + 4 * DD, DMODEL, FDIM);
    convert_transpose_kernel<<<dim3(16, 64), 256, 0, stream>>>(W2 + l * DF, base + 4 * DD + DF, FDIM, DMODEL);
  }

  embed_kernel<<<MTOT, 256, 0, stream>>>(src, tok_emb, pos_emb, X, Xb);

  const dim3 gD(DMODEL / BN, MTOT / BM);    // (8, 32)
  const dim3 gF(FDIM / BN,   MTOT / BM);    // (32, 32)
  const dim3 gA(S_LEN / 16, NHEAD, BATCH);  // (64, 16, 4)

  for (int l = 0; l < NLAYER; ++l) {
    unsigned short* base = WT + (size_t)l * LW;
    const unsigned short* wqT = base;
    const unsigned short* wkT = base + DD;
    const unsigned short* wvT = base + 2 * DD;
    const unsigned short* woT = base + 3 * DD;
    const unsigned short* w1T = base + 4 * DD;
    const unsigned short* w2T = base + 4 * DD + DF;
    const float* bq_l = bq + (size_t)l * DMODEL;
    const float* bk_l = bk + (size_t)l * DMODEL;
    const float* bv_l = bv + (size_t)l * DMODEL;
    const float* bo_l = bo + (size_t)l * DMODEL;
    const float* b1_l = b1 + (size_t)l * FDIM;
    const float* b2_l = b2 + (size_t)l * DMODEL;
    const float* rb_l = rel_bias + (size_t)l * 32 * NHEAD;
    const float* g1 = ln1_g + (size_t)l * DMODEL;
    const float* e1 = ln1_b + (size_t)l * DMODEL;
    const float* g2 = ln2_g + (size_t)l * DMODEL;
    const float* e2 = ln2_b + (size_t)l * DMODEL;

    gemm_bf16_tdm_kernel<<<gD, 256, 0, stream>>>(Xb, wqT, bq_l, T1, Qb, MTOT, DMODEL, DMODEL, 4);
    gemm_bf16_tdm_kernel<<<gD, 256, 0, stream>>>(Xb, wkT, bk_l, T1, Kb, MTOT, DMODEL, DMODEL, 4);
    gemm_bf16_tdm_kernel<<<gD, 256, 0, stream>>>(Xb, wvT, bv_l, T1, Vb, MTOT, DMODEL, DMODEL, 8);

    attn_flash_kernel<<<gA, 32, 0, stream>>>(Qb, Kb, Vb, src_mask, rb_l, AOb);

    gemm_bf16_tdm_kernel<<<gD, 256, 0, stream>>>(AOb, woT, bo_l, T1, Qb, MTOT, DMODEL, DMODEL, 2);
    ln_res_kernel<<<MTOT, 256, 0, stream>>>(X, T1, g1, e1, Xb);

    gemm_bf16_tdm_kernel<<<gF, 256, 0, stream>>>(Xb, w1T, b1_l, T1, FFb, MTOT, FDIM, DMODEL, 4 | 1);
    gemm_bf16_tdm_kernel<<<gD, 256, 0, stream>>>(FFb, w2T, b2_l, T1, Qb, MTOT, DMODEL, FDIM, 2);
    ln_res_kernel<<<MTOT, 256, 0, stream>>>(X, T1, g2, e2, Xb);
  }

  hipMemcpyAsync(d_out, X, sizeof(float) * ND, hipMemcpyDeviceToDevice, stream);
}
